// GCN_3layer_30339648979121
// MI455X (gfx1250) — compile-verified, matrix operand order checked by reading, not run on hardware
//
#include <hip/hip_runtime.h>
#include <hip/hip_bf16.h>
#include <stdint.h>

// ---------------- problem constants (match reference) ----------------
#define N_NODES 50000
#define N_EDGES 800000
#define CH      128          // IN_CH == HID == 128
#define OUT_CH  10
#define NUM_G   64
#define M_TILES (N_NODES / 16)   // 3125, exact

typedef _Float16 v16h __attribute__((ext_vector_type(16)));
typedef _Float16 v8h  __attribute__((ext_vector_type(8)));
typedef float    v8f  __attribute__((ext_vector_type(8)));

// ---------------- degree / normalization ----------------
__global__ void k_deg_init(float* __restrict__ deg) {
    int i = blockIdx.x * blockDim.x + threadIdx.x;
    if (i < N_NODES) deg[i] = 1.0f;   // self loop
}

__global__ void k_deg_edges(const int* __restrict__ dst, float* __restrict__ deg) {
    int e = blockIdx.x * blockDim.x + threadIdx.x;
    if (e < N_EDGES) atomicAdd(&deg[dst[e]], 1.0f);
}

__global__ void k_dinv(float* __restrict__ deg) {
    int i = blockIdx.x * blockDim.x + threadIdx.x;
    if (i < N_NODES) {
        float d = deg[i];
        deg[i] = (d > 0.0f) ? rsqrtf(d) : 0.0f;   // in-place -> dinv
    }
}

__global__ void k_edge_norm(const int* __restrict__ src, const int* __restrict__ dst,
                            const float* __restrict__ dinv, float* __restrict__ norm) {
    int e = blockIdx.x * blockDim.x + threadIdx.x;
    if (e < N_EDGES) norm[e] = dinv[src[e]] * dinv[dst[e]];
}

// ---------------- fp32 -> f16 conversions ----------------
__global__ void k_f32_to_f16(const float* __restrict__ in, _Float16* __restrict__ out,
                             int n, int relu) {
    int i = blockIdx.x * blockDim.x + threadIdx.x;
    if (i < n) {
        float v = in[i];
        if (relu) v = fmaxf(v, 0.0f);
        out[i] = (_Float16)v;
    }
}

// W[k][n] fp32 (row-major K x CH)  ->  Wt[n][k] f16 (row-major CH x CH)
__global__ void k_w_transpose_f16(const float* __restrict__ W, _Float16* __restrict__ Wt) {
    int k = blockIdx.x;        // 0..127
    int n = threadIdx.x;       // 0..127
    Wt[n * CH + k] = (_Float16)W[k * CH + n];
}

// ---------------- WMMA GEMM with fused self-loop + bias epilogue ----------------
// H[N,128] = Xh @ W ;  agg[N,128] = H * dinv^2 + b   (self-loop contribution init)
// one wave -> one 16x16 output tile; block of 256 threads = 8 waves = 8 N-tiles
__global__ __launch_bounds__(256) void k_gemm_wmma(const _Float16* __restrict__ Xh,
                                                   const _Float16* __restrict__ Wt,
                                                   const float* __restrict__ dinv,
                                                   const float* __restrict__ bias,
                                                   float* __restrict__ H,
                                                   float* __restrict__ agg) {
    const int wave = threadIdx.x >> 5;        // 0..7 -> n-tile
    const int lane = threadIdx.x & 31;
    const int mt   = blockIdx.x;              // 0..M_TILES-1
    const int lr   = lane & 15;
    const int hi   = lane >> 4;               // 0 or 1
    const int sel  = hi * 8;                  // K-offset split per ISA A/B f16 layout

    const size_t arow = (size_t)(mt * 16 + lr) * CH;   // A: row m, contiguous K
    const size_t brow = (size_t)(wave * 16 + lr) * CH; // B: col n of W == row n of Wt

    v8f c = {};
#pragma unroll
    for (int kk = 0; kk < 4; ++kk) {
        const int k0 = kk * 32 + sel;
        v8h alo = *(const v8h*)(Xh + arow + k0);
        v8h ahi = *(const v8h*)(Xh + arow + k0 + 16);
        v8h blo = *(const v8h*)(Wt + brow + k0);
        v8h bhi = *(const v8h*)(Wt + brow + k0 + 16);
        v16h a, b;
#pragma unroll
        for (int i = 0; i < 8; ++i) {
            a[i] = alo[i]; a[i + 8] = ahi[i];
            b[i] = blo[i]; b[i + 8] = bhi[i];
        }
        c = __builtin_amdgcn_wmma_f32_16x16x32_f16(false, a, false, b,
                                                   (short)0, c, false, false);
    }

    // D layout: VGPR v -> row v + 8*hi, col lr  (coalesced stores)
    const int ncol = wave * 16 + lr;
    const float bn = bias[ncol];
#pragma unroll
    for (int v = 0; v < 8; ++v) {
        const int row = mt * 16 + hi * 8 + v;
        const float dv = dinv[row];              // broadcast across the 16 lanes of a row
        const size_t idx = (size_t)row * CH + ncol;
        H[idx]   = c[v];
        agg[idx] = c[v] * dv * dv + bn;          // self loop + bias
    }
}

// ---------------- aggregation: edge scatter ----------------
// 32 lanes per edge, float4 per lane -> contiguous 512B gather of H[src]
__global__ void k_edge_scatter(const int* __restrict__ src, const int* __restrict__ dst,
                               const float* __restrict__ norm, const float* __restrict__ H,
                               float* __restrict__ agg) {
    int tid = blockIdx.x * blockDim.x + threadIdx.x;   // E * 32 threads
    int e = tid >> 5;
    if (e >= N_EDGES) return;
    int q = (tid & 31) << 2;                           // channel group: 4 floats
    int s = src[e], d = dst[e];
    float w = norm[e];
    const float4 h = *(const float4*)(H + (size_t)s * CH + q);
    float* ap = agg + (size_t)d * CH + q;
    atomicAdd(ap + 0, h.x * w);
    atomicAdd(ap + 1, h.y * w);
    atomicAdd(ap + 2, h.z * w);
    atomicAdd(ap + 3, h.w * w);
}

// ---------------- pooling + head ----------------
__global__ void k_pool_zero(float* __restrict__ gsum, float* __restrict__ gcnt) {
    int i = blockIdx.x * blockDim.x + threadIdx.x;
    if (i < NUM_G * CH) gsum[i] = 0.0f;
    if (i < NUM_G) gcnt[i] = 0.0f;
}

__global__ void k_pool_scatter(const float* __restrict__ nemb, const int* __restrict__ batch,
                               float* __restrict__ gsum, float* __restrict__ gcnt) {
    int i = blockIdx.x * blockDim.x + threadIdx.x;     // over N_NODES*CH
    if (i < N_NODES * CH) {
        int node = i >> 7;
        int c = i & (CH - 1);
        int g = batch[node];
        atomicAdd(&gsum[g * CH + c], nemb[i]);
        if (c == 0) atomicAdd(&gcnt[g], 1.0f);
    }
}

__global__ void k_final_linear(const float* __restrict__ gsum, const float* __restrict__ gcnt,
                               const float* __restrict__ Wl, const float* __restrict__ bl,
                               float* __restrict__ out) {
    int g = blockIdx.x;        // 0..63
    int o = threadIdx.x;       // 0..15, guard to 10
    if (o >= OUT_CH) return;
    float inv = 1.0f / fmaxf(gcnt[g], 1.0f);
    float acc = bl[o];
    for (int k = 0; k < CH; ++k)
        acc += (gsum[g * CH + k] * inv) * Wl[k * OUT_CH + o];
    out[g * OUT_CH + o] = acc;
}

// ---------------- host-side orchestration ----------------
static inline size_t align256(size_t x) { return (x + 255) & ~(size_t)255; }

extern "C" void kernel_launch(void* const* d_in, const int* in_sizes, int n_in,
                              void* d_out, int out_size, void* d_ws, size_t ws_size,
                              hipStream_t stream) {
    const float* x     = (const float*)d_in[0];
    const int*   eidx  = (const int*)d_in[1];
    const int*   batch = (const int*)d_in[2];
    const float* W1 = (const float*)d_in[3];
    const float* b1 = (const float*)d_in[4];
    const float* W2 = (const float*)d_in[5];
    const float* b2 = (const float*)d_in[6];
    const float* W3 = (const float*)d_in[7];
    const float* b3 = (const float*)d_in[8];
    const float* Wl = (const float*)d_in[9];
    const float* bl = (const float*)d_in[10];

    const int* src = eidx;               // edge_index[0]
    const int* dst = eidx + N_EDGES;     // edge_index[1]

    float* g_emb = (float*)d_out;                  // [64,10]
    float* n_emb = (float*)d_out + NUM_G * OUT_CH; // [N,128] (layer-3 agg target)

    // workspace carve-up
    char* ws = (char*)d_ws;
    size_t off = 0;
    float*    dinv = (float*)(ws + off);    off = align256(off + (size_t)N_NODES * 4);
    float*    norm = (float*)(ws + off);    off = align256(off + (size_t)N_EDGES * 4);
    _Float16* Xh   = (_Float16*)(ws + off); off = align256(off + (size_t)N_NODES * CH * 2);
    _Float16* Wt   = (_Float16*)(ws + off); off = align256(off + (size_t)CH * CH * 2);
    float*    Hbuf = (float*)(ws + off);    off = align256(off + (size_t)N_NODES * CH * 4);
    float*    aggA = (float*)(ws + off);    off = align256(off + (size_t)N_NODES * CH * 4);
    float*    gsum = (float*)(ws + off);    off = align256(off + (size_t)NUM_G * CH * 4);
    float*    gcnt = (float*)(ws + off);    off = align256(off + (size_t)NUM_G * 4);
    (void)ws_size; (void)n_in; (void)in_sizes; (void)out_size;

    const int T = 256;
    const int NC = N_NODES * CH;                      // 6.4M
    const int gridNodes = (N_NODES + T - 1) / T;
    const int gridEdges = (N_EDGES + T - 1) / T;
    const int gridNC    = (NC + T - 1) / T;
    const int gridScat  = (N_EDGES * 32 + T - 1) / T; // 100000 blocks

    // degree / norms (reused for all 3 layers)
    k_deg_init <<<gridNodes, T, 0, stream>>>(dinv);
    k_deg_edges<<<gridEdges, T, 0, stream>>>(dst, dinv);
    k_dinv     <<<gridNodes, T, 0, stream>>>(dinv);
    k_edge_norm<<<gridEdges, T, 0, stream>>>(src, dst, dinv, norm);

    struct Layer { const float* in; const float* W; const float* b; float* out; int relu_in; };
    Layer layers[3] = {
        { x,    W1, b1, aggA,  0 },   // layer 1: input x (no relu)
        { aggA, W2, b2, aggA,  1 },   // layer 2: relu(agg1); Xh conversion precedes overwrite
        { aggA, W3, b3, n_emb, 1 },   // layer 3: relu(agg2) -> n_emb (no output relu)
    };

    for (int l = 0; l < 3; ++l) {
        k_f32_to_f16     <<<gridNC, T, 0, stream>>>(layers[l].in, Xh, NC, layers[l].relu_in);
        k_w_transpose_f16<<<CH, CH, 0, stream>>>(layers[l].W, Wt);
        // GEMM with fused self-loop + bias init of the aggregation buffer
        k_gemm_wmma      <<<M_TILES, 256, 0, stream>>>(Xh, Wt, dinv, layers[l].b,
                                                       Hbuf, layers[l].out);
        k_edge_scatter   <<<gridScat, T, 0, stream>>>(src, dst, norm, Hbuf, layers[l].out);
    }

    // global mean pool + linear head
    k_pool_zero   <<<(NUM_G * CH + T - 1) / T, T, 0, stream>>>(gsum, gcnt);
    k_pool_scatter<<<gridNC, T, 0, stream>>>(n_emb, batch, gsum, gcnt);
    k_final_linear<<<NUM_G, 16, 0, stream>>>(gsum, gcnt, Wl, bl, g_emb);
}
